// CaPa_MoE_without_clinical_31379031065168
// MI455X (gfx1250) — matrix operand order repaired and checked by reading, not compiled, not run on hardware
//
#include <hip/hip_runtime.h>
#include <cstdint>
#include <cstddef>

typedef __bf16 bf16;
typedef __attribute__((ext_vector_type(16))) __bf16 v16bf;
typedef __attribute__((ext_vector_type(8)))  __bf16 v8bf;
typedef __attribute__((ext_vector_type(8)))  float  v8f;

__device__ __forceinline__ v8f splat8(float x) {
  v8f v = {x, x, x, x, x, x, x, x};
  return v;
}

// ---------------------------------------------------------------------------
// Pack a row-major fp32 weight [K,N] into bf16 WMMA B-operand tile layout.
// Tile (kt, nt) covers K rows [kt*32, kt*32+32), N cols [nt*16, nt*16+16).
// Lane L (0..31): column nt*16 + (L&15), K = kt*32 + ((L>>4)<<3) + {0..7, 16..23},
// 16 contiguous bf16 per lane -> one 32B load in the GEMM inner loop.
// ---------------------------------------------------------------------------
__global__ void pack_b_kernel(const float* __restrict__ B, bf16* __restrict__ Bp,
                              int K, int N) {
  int idx = blockIdx.x * 256 + threadIdx.x;
  int total = (K >> 5) * (N >> 4) * 32;
  if (idx >= total) return;
  int lane = idx & 31;
  int tile = idx >> 5;
  int ntiles = N >> 4;
  int nt = tile % ntiles;
  int kt = tile / ntiles;
  int col = nt * 16 + (lane & 15);
  int kb  = kt * 32 + ((lane >> 4) << 3);
  bf16* dst = Bp + (size_t)idx * 16;
#pragma unroll
  for (int h = 0; h < 16; ++h)
    dst[h] = (bf16)B[(size_t)(kb + h + ((h >> 3) << 3)) * N + col];
}

// ---------------------------------------------------------------------------
// WMMA GEMM:  C[M,N] = act(A[M,K] @ B[K,N] + bias)  (C stored bf16 row-major)
// A either fp32 (converted in-register) or bf16, row-major.
// Block = 256 threads = 8 waves; each wave owns a 16x64 output strip
// (4 x 16x16 accumulators), single K loop in steps of 32 with
// v_wmma_f32_16x16x32_bf16 -> nothing loop-invariant to hoist/spill.
// ---------------------------------------------------------------------------
template <bool A_BF16, bool RELU>
__global__ void gemm_wmma_kernel(const void* __restrict__ Av,
                                 const bf16* __restrict__ Bp,
                                 const float* __restrict__ bias,
                                 bf16* __restrict__ C,
                                 int M, int N, int K) {
  const int lane = threadIdx.x & 31;
  const int gw   = blockIdx.x * 8 + (threadIdx.x >> 5);
  const int nblk = N >> 6;            // 64-wide strips
  const int mt   = gw / nblk;
  const int nb   = gw % nblk;
  if (mt >= (M >> 4)) return;         // wave-uniform exit, EXEC stays full
  const int row  = mt * 16 + (lane & 15);
  const int koff = (lane >> 4) << 3;  // A-operand half selector
  v8f acc[4];
#pragma unroll
  for (int t = 0; t < 4; ++t)
    acc[t] = splat8(bias[nb * 64 + t * 16 + (lane & 15)]);
  const int KT = K >> 5;
  const int ntiles = N >> 4;
  for (int kt = 0; kt < KT; ++kt) {
    v16bf a;
    if (A_BF16) {
      const bf16* ap = (const bf16*)Av + (size_t)row * K + kt * 32 + koff;
      v8bf lo = *(const v8bf*)(ap);
      v8bf hi = *(const v8bf*)(ap + 16);
      a = __builtin_shufflevector(lo, hi, 0, 1, 2, 3, 4, 5, 6, 7,
                                  8, 9, 10, 11, 12, 13, 14, 15);
    } else {
      const float* ap = (const float*)Av + (size_t)row * K + kt * 32 + koff;
      float4 x0 = *(const float4*)(ap);
      float4 x1 = *(const float4*)(ap + 4);
      float4 y0 = *(const float4*)(ap + 16);
      float4 y1 = *(const float4*)(ap + 20);
      a[0]  = (bf16)x0.x; a[1]  = (bf16)x0.y; a[2]  = (bf16)x0.z; a[3]  = (bf16)x0.w;
      a[4]  = (bf16)x1.x; a[5]  = (bf16)x1.y; a[6]  = (bf16)x1.z; a[7]  = (bf16)x1.w;
      a[8]  = (bf16)y0.x; a[9]  = (bf16)y0.y; a[10] = (bf16)y0.z; a[11] = (bf16)y0.w;
      a[12] = (bf16)y1.x; a[13] = (bf16)y1.y; a[14] = (bf16)y1.z; a[15] = (bf16)y1.w;
    }
#pragma unroll
    for (int t = 0; t < 4; ++t) {
      const bf16* bp = Bp + (((size_t)kt * ntiles + (nb * 4 + t)) * 32 + lane) * 16;
      v16bf b = *(const v16bf*)bp;
      acc[t] = __builtin_amdgcn_wmma_f32_16x16x32_bf16(
          false, a, false, b, (short)0, acc[t], false, false);
    }
  }
  const int crow = mt * 16 + ((lane >> 4) << 3);
#pragma unroll
  for (int t = 0; t < 4; ++t) {
    int col = nb * 64 + t * 16 + (lane & 15);
#pragma unroll
    for (int r = 0; r < 8; ++r) {
      float v = acc[t][r];
      if (RELU) v = fmaxf(v, 0.0f);
      C[(size_t)(crow + r) * N + col] = (bf16)v;
    }
  }
}

// ---------------------------------------------------------------------------
// Gated-attention GEMM pass:
//   G = tanh(hp@Wa + ba) * sigmoid(hp@Wb + bb)      [M, 256] bf16
// Same single-K-loop skeleton as gemm_wmma_kernel, but dual B operands
// sharing one A operand (8 WMMAs per kt for a 16x64 strip). No nested
// loop-invariant loads -> no scratch spill of A tiles.
// ---------------------------------------------------------------------------
__global__ void gated_gemm_kernel(const bf16* __restrict__ A,    // [M,512] bf16
                                  const bf16* __restrict__ BaP,  // packed 512x256
                                  const bf16* __restrict__ BbP,  // packed 512x256
                                  const float* __restrict__ ba,
                                  const float* __restrict__ bb,
                                  bf16* __restrict__ G,          // [M,256] bf16
                                  int M) {
  const int N = 256, K = 512;
  const int lane = threadIdx.x & 31;
  const int gw   = blockIdx.x * 8 + (threadIdx.x >> 5);
  const int nblk = N >> 6;                 // 4 strips of 64
  const int mt   = gw / nblk;
  const int nb   = gw % nblk;
  if (mt >= (M >> 4)) return;
  const int row  = mt * 16 + (lane & 15);
  const int koff = (lane >> 4) << 3;
  v8f accA[4], accB[4];
#pragma unroll
  for (int t = 0; t < 4; ++t) {
    accA[t] = splat8(ba[nb * 64 + t * 16 + (lane & 15)]);
    accB[t] = splat8(bb[nb * 64 + t * 16 + (lane & 15)]);
  }
  const int ntiles = N >> 4;
  for (int kt = 0; kt < (K >> 5); ++kt) {
    const bf16* ap = A + (size_t)row * K + kt * 32 + koff;
    v8bf lo = *(const v8bf*)(ap);
    v8bf hi = *(const v8bf*)(ap + 16);
    v16bf a = __builtin_shufflevector(lo, hi, 0, 1, 2, 3, 4, 5, 6, 7,
                                      8, 9, 10, 11, 12, 13, 14, 15);
#pragma unroll
    for (int t = 0; t < 4; ++t) {
      size_t tb = (((size_t)kt * ntiles + (nb * 4 + t)) * 32 + lane) * 16;
      v16bf bA = *(const v16bf*)(BaP + tb);
      v16bf bB = *(const v16bf*)(BbP + tb);
      accA[t] = __builtin_amdgcn_wmma_f32_16x16x32_bf16(
          false, a, false, bA, (short)0, accA[t], false, false);
      accB[t] = __builtin_amdgcn_wmma_f32_16x16x32_bf16(
          false, a, false, bB, (short)0, accB[t], false, false);
    }
  }
  const int crow = mt * 16 + ((lane >> 4) << 3);
#pragma unroll
  for (int t = 0; t < 4; ++t) {
    int col = nb * 64 + t * 16 + (lane & 15);
#pragma unroll
    for (int r = 0; r < 8; ++r) {
      float x  = tanhf(accA[t][r]);
      float y  = 1.0f / (1.0f + expf(-accB[t][r]));
      G[(size_t)(crow + r) * N + col] = (bf16)(x * y);
    }
  }
}

// ---------------------------------------------------------------------------
// scores[c, row] = sum_k G[row,k] * Wc[k,c] + bc[c]    (K=256, C=2)
// One thread per row; Wc is a tiny L2-hot broadcast.
// ---------------------------------------------------------------------------
__global__ void attn_project_kernel(const bf16* __restrict__ G,   // [M,256]
                                    const float* __restrict__ Wc, // [256,2]
                                    const float* __restrict__ bc, // [2]
                                    float* __restrict__ scores,   // [2,M]
                                    int M) {
  int row = blockIdx.x * 256 + threadIdx.x;
  if (row >= M) return;
  const bf16* g = G + (size_t)row * 256;
  float s0 = 0.f, s1 = 0.f;
#pragma unroll 4
  for (int k = 0; k < 256; ++k) {
    float gv = (float)g[k];
    s0 += gv * Wc[k * 2 + 0];
    s1 += gv * Wc[k * 2 + 1];
  }
  scores[row]     = s0 + bc[0];
  scores[M + row] = s1 + bc[1];
}

// ---------------------------------------------------------------------------
// Per-class softmax statistics (max, sum of exp) over the instance axis.
// ---------------------------------------------------------------------------
__global__ void softmax_stats_kernel(const float* __restrict__ scores,  // [2, N]
                                     float* __restrict__ stats,         // {m0,s0,m1,s1}
                                     int Ntot) {
  __shared__ float red[256];
  const int t = threadIdx.x;
  for (int c = 0; c < 2; ++c) {
    const float* s = scores + (size_t)c * Ntot;
    float mx = -3.4e38f;
    for (int i = t; i < Ntot; i += 256) mx = fmaxf(mx, s[i]);
    red[t] = mx;
    __syncthreads();
    for (int k = 128; k > 0; k >>= 1) {
      if (t < k) red[t] = fmaxf(red[t], red[t + k]);
      __syncthreads();
    }
    float m = red[0];
    __syncthreads();
    float sum = 0.f;
    for (int i = t; i < Ntot; i += 256) sum += expf(s[i] - m);
    red[t] = sum;
    __syncthreads();
    for (int k = 128; k > 0; k >>= 1) {
      if (t < k) red[t] += red[t + k];
      __syncthreads();
    }
    if (t == 0) { stats[c * 2] = m; stats[c * 2 + 1] = red[0]; }
    __syncthreads();
  }
}

// ---------------------------------------------------------------------------
// Partial softmax-weighted pooling: M[c,:] = sum_n softmax(A)[c,n] * hp[n,:]
// Each block covers a disjoint n-range; 512 threads = 512 feature columns.
// Deterministic (no atomics); partials reduced in head_kernel.
// ---------------------------------------------------------------------------
__global__ void pool_kernel(const bf16* __restrict__ hp,     // [N,512] bf16
                            const float* __restrict__ scores,// [2,N]
                            const float* __restrict__ stats, // {m0,s0,m1,s1}
                            float* __restrict__ part,        // [P,2,512]
                            int Ntot, int nPerBlk) {
  const int p = blockIdx.x, t = threadIdx.x;
  const float mx0 = stats[0], inv0 = 1.0f / stats[1];
  const float mx1 = stats[2], inv1 = 1.0f / stats[3];
  int n0 = p * nPerBlk;
  int n1 = n0 + nPerBlk;
  if (n1 > Ntot) n1 = Ntot;
  float a0 = 0.f, a1 = 0.f;
  for (int n = n0; n < n1; ++n) {
    float w0 = expf(scores[n] - mx0) * inv0;
    float w1 = expf(scores[Ntot + n] - mx1) * inv1;
    float h  = (float)hp[(size_t)n * 512 + t];
    a0 += w0 * h;
    a1 += w1 * h;
  }
  part[((size_t)p * 2 + 0) * 512 + t] = a0;
  part[((size_t)p * 2 + 1) * 512 + t] = a1;
}

// ---------------------------------------------------------------------------
// Head: reduce pooling partials -> M_v/M_u [2,512], run experts + gating +
// classifier entirely in one block (work is ~26 MFLOP, latency-bound).
// ---------------------------------------------------------------------------
__global__ void head_kernel(
    const float* __restrict__ partV, const float* __restrict__ partU, int P,
    const float* __restrict__ e1W1, const float* __restrict__ e1b1,
    const float* __restrict__ e1W2, const float* __restrict__ e1b2,
    const float* __restrict__ e3W1, const float* __restrict__ e3b1,
    const float* __restrict__ e3W2, const float* __restrict__ e3b2,
    const float* __restrict__ e2W1, const float* __restrict__ e2b1,
    const float* __restrict__ e2W2, const float* __restrict__ e2b2,
    const float* __restrict__ e2pW, const float* __restrict__ e2pb,
    const float* __restrict__ gW1,  const float* __restrict__ gb1,
    const float* __restrict__ gW2,  const float* __restrict__ gb2,
    const float* __restrict__ clsW, const float* __restrict__ clsb,
    float* __restrict__ out) {
  __shared__ float sm[12800];
  float* Mv   = sm;            // 2x512
  float* Mu   = sm + 1024;     // 2x512
  float* g    = sm + 2048;     // 1024
  float* gh   = sm + 3072;     // 256
  float* pp   = sm + 3328;     // 4
  float* F1   = sm + 3332;     // 2x512
  float* F2   = sm + 4356;     // 2x512
  float* F3   = sm + 5380;     // 2x512
  float* bufA = sm + 6404;     // 2x1024
  float* bufB = sm + 8452;     // 2x2048
  const int t = threadIdx.x;   // 512 threads

  for (int c = 0; c < 2; ++c) {
    float sv = 0.f, su = 0.f;
    for (int p = 0; p < P; ++p) {
      sv += partV[((size_t)p * 2 + c) * 512 + t];
      su += partU[((size_t)p * 2 + c) * 512 + t];
    }
    Mv[c * 512 + t] = sv;
    Mu[c * 512 + t] = su;
  }
  __syncthreads();
  // gating input: mean over classes, concat(v, u)
  g[t]       = 0.5f * (Mv[t] + Mv[512 + t]);
  g[512 + t] = 0.5f * (Mu[t] + Mu[512 + t]);
  __syncthreads();
  if (t < 256) {
    float a = gb1[t];
    for (int k = 0; k < 1024; ++k) a += g[k] * gW1[(size_t)k * 256 + t];
    gh[t] = fmaxf(a, 0.f);
  }
  __syncthreads();
  if (t < 3) {
    float a = gb2[t];
    for (int k = 0; k < 256; ++k) a += gh[k] * gW2[(size_t)k * 3 + t];
    pp[t] = a;
  }
  __syncthreads();
  if (t == 0) {
    float m = fmaxf(pp[0], fmaxf(pp[1], pp[2]));
    float e0 = expf(pp[0] - m), e1 = expf(pp[1] - m), e2 = expf(pp[2] - m);
    float s = e0 + e1 + e2;
    pp[0] = e0 / s; pp[1] = e1 / s; pp[2] = e2 / s;
  }
  // expert 1: relu(relu(Mv@e1W1+b)@e1W2+b)
  for (int c = 0; c < 2; ++c)
    for (int j = t; j < 1024; j += 512) {
      float a = e1b1[j];
      for (int k = 0; k < 512; ++k) a += Mv[c * 512 + k] * e1W1[(size_t)k * 1024 + j];
      bufA[c * 1024 + j] = fmaxf(a, 0.f);
    }
  __syncthreads();
  for (int c = 0; c < 2; ++c) {
    float a = e1b2[t];
    for (int k = 0; k < 1024; ++k) a += bufA[c * 1024 + k] * e1W2[(size_t)k * 512 + t];
    F1[c * 512 + t] = fmaxf(a, 0.f);
  }
  __syncthreads();
  // expert 3 on Mu
  for (int c = 0; c < 2; ++c)
    for (int j = t; j < 1024; j += 512) {
      float a = e3b1[j];
      for (int k = 0; k < 512; ++k) a += Mu[c * 512 + k] * e3W1[(size_t)k * 1024 + j];
      bufA[c * 1024 + j] = fmaxf(a, 0.f);
    }
  __syncthreads();
  for (int c = 0; c < 2; ++c) {
    float a = e3b2[t];
    for (int k = 0; k < 1024; ++k) a += bufA[c * 1024 + k] * e3W2[(size_t)k * 512 + t];
    F3[c * 512 + t] = fmaxf(a, 0.f);
  }
  __syncthreads();
  // expert 2 on concat(Mv, Mu) with projection
  for (int c = 0; c < 2; ++c)
    for (int j = t; j < 2048; j += 512) {
      float a = e2b1[j];
      for (int k = 0; k < 512; ++k) a += Mv[c * 512 + k] * e2W1[(size_t)k * 2048 + j];
      for (int k = 0; k < 512; ++k) a += Mu[c * 512 + k] * e2W1[(size_t)(512 + k) * 2048 + j];
      bufB[c * 2048 + j] = fmaxf(a, 0.f);
    }
  __syncthreads();
  for (int c = 0; c < 2; ++c)
    for (int j = t; j < 1024; j += 512) {
      float a = e2b2[j];
      for (int k = 0; k < 2048; ++k) a += bufB[c * 2048 + k] * e2W2[(size_t)k * 1024 + j];
      bufA[c * 1024 + j] = fmaxf(a, 0.f);
    }
  __syncthreads();
  for (int c = 0; c < 2; ++c) {
    float a = e2pb[t];
    for (int k = 0; k < 1024; ++k) a += bufA[c * 1024 + k] * e2pW[(size_t)k * 512 + t];
    F2[c * 512 + t] = a;
  }
  __syncthreads();
  // fused logits: logit[c] = sum_h fused[c,h]*clsW[c,h] + clsb[c]
  const float p0 = pp[0], p1 = pp[1], p2 = pp[2];
  bufB[t]       = (p0 * F1[t]       + p1 * F2[t]       + p2 * F3[t])       * clsW[t];
  bufB[512 + t] = (p0 * F1[512 + t] + p1 * F2[512 + t] + p2 * F3[512 + t]) * clsW[512 + t];
  __syncthreads();
  for (int s = 256; s > 0; s >>= 1) {
    if (t < s) { bufB[t] += bufB[t + s]; bufB[512 + t] += bufB[512 + t + s]; }
    __syncthreads();
  }
  if (t == 0) {
    float L0 = bufB[0] + clsb[0];
    float L1 = bufB[512] + clsb[1];
    out[0] = L0;
    out[1] = L1;
    float m = fmaxf(L0, L1);
    float e0 = expf(L0 - m), e1 = expf(L1 - m), s = e0 + e1;
    out[2] = e0 / s;
    out[3] = e1 / s;
    out[4] = (L1 > L0) ? 1.0f : 0.0f;  // argmax (ties -> 0, matching jnp.argmax)
  }
}

// ---------------------------------------------------------------------------
extern "C" void kernel_launch(void* const* d_in, const int* in_sizes, int n_in,
                              void* d_out, int out_size, void* d_ws, size_t ws_size,
                              hipStream_t stream) {
  (void)in_sizes; (void)n_in; (void)out_size; (void)ws_size;
  const float* h_vir = (const float*)d_in[0];
  const float* h_uni = (const float*)d_in[1];
  const float* Wp  = (const float*)d_in[3];
  const float* bp  = (const float*)d_in[4];
  const float* vW1 = (const float*)d_in[5];  const float* vb1 = (const float*)d_in[6];
  const float* vWa = (const float*)d_in[7];  const float* vba = (const float*)d_in[8];
  const float* vWb = (const float*)d_in[9];  const float* vbb = (const float*)d_in[10];
  const float* vWc = (const float*)d_in[11]; const float* vbc = (const float*)d_in[12];
  const float* uW1 = (const float*)d_in[13]; const float* ub1 = (const float*)d_in[14];
  const float* uWa = (const float*)d_in[15]; const float* uba = (const float*)d_in[16];
  const float* uWb = (const float*)d_in[17]; const float* ubb = (const float*)d_in[18];
  const float* uWc = (const float*)d_in[19]; const float* ubc = (const float*)d_in[20];
  const float* e1W1 = (const float*)d_in[21]; const float* e1b1 = (const float*)d_in[22];
  const float* e1W2 = (const float*)d_in[23]; const float* e1b2 = (const float*)d_in[24];
  const float* e2W1 = (const float*)d_in[25]; const float* e2b1 = (const float*)d_in[26];
  const float* e2W2 = (const float*)d_in[27]; const float* e2b2 = (const float*)d_in[28];
  const float* e2pW = (const float*)d_in[29]; const float* e2pb = (const float*)d_in[30];
  const float* e3W1 = (const float*)d_in[31]; const float* e3b1 = (const float*)d_in[32];
  const float* e3W2 = (const float*)d_in[33]; const float* e3b2 = (const float*)d_in[34];
  const float* gW1  = (const float*)d_in[35]; const float* gb1  = (const float*)d_in[36];
  const float* gW2  = (const float*)d_in[37]; const float* gb2  = (const float*)d_in[38];
  const float* clsW = (const float*)d_in[39]; const float* clsb = (const float*)d_in[40];
  float* out = (float*)d_out;

  // workspace carve (256B aligned)
  char* ws = (char*)d_ws;
  size_t off = 0;
  auto carve = [&](size_t bytes) -> char* {
    char* p = ws + off;
    off = (off + bytes + 255) & ~(size_t)255;
    return p;
  };
  const int M = 30000;
  bf16* WpP  = (bf16*)carve(2560ull * 1024 * 2);
  bf16* vW1P = (bf16*)carve(1024ull * 512 * 2);
  bf16* uW1P = (bf16*)carve(1024ull * 512 * 2);
  bf16* vWaP = (bf16*)carve(512ull * 256 * 2);
  bf16* vWbP = (bf16*)carve(512ull * 256 * 2);
  bf16* uWaP = (bf16*)carve(512ull * 256 * 2);
  bf16* uWbP = (bf16*)carve(512ull * 256 * 2);
  bf16* hvB  = (bf16*)carve((size_t)M * 1024 * 2);
  bf16* hpvB = (bf16*)carve((size_t)M * 512 * 2);
  bf16* hpuB = (bf16*)carve((size_t)M * 512 * 2);
  bf16* Gv   = (bf16*)carve((size_t)M * 256 * 2);
  bf16* Gu   = (bf16*)carve((size_t)M * 256 * 2);
  float* statsV = (float*)carve(4 * sizeof(float));
  float* statsU = (float*)carve(4 * sizeof(float));
  const int P = 120;
  float* partV = (float*)carve((size_t)P * 2 * 512 * sizeof(float));
  float* partU = (float*)carve((size_t)P * 2 * 512 * sizeof(float));

  auto packLaunch = [&](const float* src, bf16* dst, int K, int N) {
    int total = (K >> 5) * (N >> 4) * 32;
    pack_b_kernel<<<(total + 255) / 256, 256, 0, stream>>>(src, dst, K, N);
  };
  packLaunch(Wp,  WpP,  2560, 1024);
  packLaunch(vW1, vW1P, 1024, 512);
  packLaunch(uW1, uW1P, 1024, 512);
  packLaunch(vWa, vWaP, 512, 256);
  packLaunch(vWb, vWbP, 512, 256);
  packLaunch(uWa, uWaP, 512, 256);
  packLaunch(uWb, uWbP, 512, 256);

  {  // hv = h_virchow @ Wp + bp      (fp32 A, no act, bf16 out)
    int waves = (M / 16) * (1024 / 64);
    gemm_wmma_kernel<false, false><<<(waves + 7) / 8, 256, 0, stream>>>(
        h_vir, WpP, bp, hvB, M, 1024, 2560);
  }
  {  // hp_v = relu(hv @ vW1 + vb1)   (bf16 A)
    int waves = (M / 16) * (512 / 64);
    gemm_wmma_kernel<true, true><<<(waves + 7) / 8, 256, 0, stream>>>(
        hvB, vW1P, vb1, hpvB, M, 512, 1024);
  }
  {  // hp_u = relu(h_UNI @ uW1 + ub1) (fp32 A)
    int waves = (M / 16) * (512 / 64);
    gemm_wmma_kernel<false, true><<<(waves + 7) / 8, 256, 0, stream>>>(
        h_uni, uW1P, ub1, hpuB, M, 512, 1024);
  }

  {  // gated attention hidden: G = tanh(hp@Wa+ba)*sigmoid(hp@Wb+bb)
    int waves = (M / 16) * (256 / 64);
    int blocks = (waves + 7) / 8;
    gated_gemm_kernel<<<blocks, 256, 0, stream>>>(hpvB, vWaP, vWbP, vba, vbb, Gv, M);
    gated_gemm_kernel<<<blocks, 256, 0, stream>>>(hpuB, uWaP, uWbP, uba, ubb, Gu, M);
  }

  // raw attention scores, written straight into the output tensor slots
  float* scoresV = out + 5;
  float* scoresU = out + 5 + 2 * M;
  {
    int blocks = (M + 255) / 256;
    attn_project_kernel<<<blocks, 256, 0, stream>>>(Gv, vWc, vbc, scoresV, M);
    attn_project_kernel<<<blocks, 256, 0, stream>>>(Gu, uWc, ubc, scoresU, M);
  }
  softmax_stats_kernel<<<1, 256, 0, stream>>>(scoresV, statsV, M);
  softmax_stats_kernel<<<1, 256, 0, stream>>>(scoresU, statsU, M);

  const int nPerBlk = (M + P - 1) / P;
  pool_kernel<<<P, 512, 0, stream>>>(hpvB, scoresV, statsV, partV, M, nPerBlk);
  pool_kernel<<<P, 512, 0, stream>>>(hpuB, scoresU, statsU, partU, M, nPerBlk);

  head_kernel<<<1, 512, 0, stream>>>(
      partV, partU, P,
      e1W1, e1b1, e1W2, e1b2,
      e3W1, e3b1, e3W2, e3b2,
      e2W1, e2b1, e2W2, e2b2, e2pW, e2pb,
      gW1, gb1, gW2, gb2, clsW, clsb, out);
}
